// MSAColumnAttentionV44_29197187678431
// MI455X (gfx1250) — matrix operand order repaired and checked by reading, compiled, not run on hardware
//
#include <hip/hip_runtime.h>
#include <hip/hip_bf16.h>

typedef __attribute__((ext_vector_type(16))) __bf16 v16bf;
typedef __attribute__((ext_vector_type(8)))  float  v8f;

#define C_DIM 256
#define S_DIM 192
#define N_DIM 256
#define H_DIM 8
#define D_DIM 32

__device__ __forceinline__ v8f wmma_bf16(v16bf a, v16bf b, v8f c) {
  return __builtin_amdgcn_wmma_f32_16x16x32_bf16(
      /*neg_a=*/false, a, /*neg_b=*/false, b,
      /*c_mod=*/(short)0, c, /*reuse_a=*/false, /*reuse_b=*/false);
}

// A fragment (16x32 bf16), gathered from an fp32 row-major source with leading dim ld.
// Lane holds row (lane&15); K chunks [koff, koff+8) and [koff+16, koff+24), koff=(lane>>4)*8.
__device__ __forceinline__ v16bf load_a_f32(const float* __restrict__ base, int ld, int lane) {
  const int row  = lane & 15;
  const int koff = (lane >> 4) << 3;
  const float* p = base + row * ld + koff;
  v16bf a;
#pragma unroll
  for (int i = 0; i < 8; ++i) a[i]     = (__bf16)p[i];
#pragma unroll
  for (int i = 0; i < 8; ++i) a[8 + i] = (__bf16)p[16 + i];
  return a;
}

// A fragment from a bf16 row-major source.
__device__ __forceinline__ v16bf load_a_bf16(const __bf16* __restrict__ base, int ld, int lane) {
  const int row  = lane & 15;
  const int koff = (lane >> 4) << 3;
  const __bf16* p = base + row * ld + koff;
  v16bf a;
#pragma unroll
  for (int i = 0; i < 8; ++i) a[i]     = p[i];
#pragma unroll
  for (int i = 0; i < 8; ++i) a[8 + i] = p[16 + i];
  return a;
}

// B fragment (32x16 bf16), B[k][j] = W[j][k] with W row-major (ld = row stride).
// Lane holds column (lane&15); K chunks [koff, koff+8) and [koff+16, koff+24), koff=(lane>>4)*8.
__device__ __forceinline__ v16bf load_b_bf16(const __bf16* __restrict__ base, int ld, int lane) {
  const int col  = lane & 15;
  const int koff = (lane >> 4) << 3;
  const __bf16* p = base + col * ld + koff;
  v16bf b;
#pragma unroll
  for (int i = 0; i < 8; ++i) b[i]     = p[i];
#pragma unroll
  for (int i = 0; i < 8; ++i) b[8 + i] = p[16 + i];
  return b;
}

// ---------------- Kernel 0: weight conversion fp32 -> bf16 ----------------
__global__ void convert_weights_kernel(const float* __restrict__ wq, const float* __restrict__ wo,
                                       __bf16* __restrict__ wq_bf, __bf16* __restrict__ wo_bf) {
  int i = blockIdx.x * blockDim.x + threadIdx.x;
  if (i < 3 * C_DIM * C_DIM) wq_bf[i] = (__bf16)wq[i];
  if (i < C_DIM * C_DIM)     wo_bf[i] = (__bf16)wo[i];
}

// ---------------- Kernel 1: QKV projection GEMM ----------------
// Per n: Q/K/V = msa_t[n] (192x256) @ w_qkv^T (256x768) + b_qkv.
// Grid: 256 n * 12 row-tiles. Block: 256 threads (8 waves), each wave owns 6 col-tiles (96 cols).
__global__ void __launch_bounds__(256) qkv_kernel(
    const float* __restrict__ msa, const __bf16* __restrict__ wq, const float* __restrict__ bq,
    __bf16* __restrict__ Qb, __bf16* __restrict__ Kb, __bf16* __restrict__ Vt) {
  const int n    = blockIdx.x / 12;
  const int s0   = (blockIdx.x % 12) * 16;
  const int wave = threadIdx.x >> 5;
  const int lane = threadIdx.x & 31;

  v8f acc[6] = {};
  const float* abase = msa + (s0 * N_DIM + n) * C_DIM;  // row stride over s = N*C fp32

#pragma unroll 2
  for (int k0 = 0; k0 < C_DIM; k0 += 32) {
    // Issue all loads of this K-step first so they form one clause and the
    // WMMAs can retire against decreasing loadcnt instead of waiting to 0.
    v16bf a = load_a_f32(abase + k0, N_DIM * C_DIM, lane);
    if (k0 + 32 < C_DIM)
      __builtin_prefetch(abase + k0 + 32 + (lane & 15) * (N_DIM * C_DIM), 0, 1);
    v16bf b[6];
#pragma unroll
    for (int i = 0; i < 6; ++i)
      b[i] = load_b_bf16(wq + ((wave * 6 + i) * 16) * C_DIM + k0, C_DIM, lane);
#pragma unroll
    for (int i = 0; i < 6; ++i)
      acc[i] = wmma_bf16(a, b[i], acc[i]);
  }

  const int rbase = (lane >> 4) << 3;
  const int cl    = lane & 15;
#pragma unroll
  for (int i = 0; i < 6; ++i) {
    const int j     = (wave * 6 + i) * 16 + cl;   // 0..767
    const int three = j >> 8;
    const int h     = (j >> 5) & 7;
    const int d     = j & 31;
    const float bias = bq[j];
#pragma unroll
    for (int v = 0; v < 8; ++v) {
      const int s = s0 + rbase + v;
      const __bf16 bv = (__bf16)(acc[i][v] + bias);
      if (three == 0)      Qb[((n * H_DIM + h) * S_DIM + s) * D_DIM + d] = bv;
      else if (three == 1) Kb[((n * H_DIM + h) * S_DIM + s) * D_DIM + d] = bv;
      else                 Vt[((n * H_DIM + h) * D_DIM + d) * S_DIM + s] = bv;  // V transposed
    }
  }
}

// ---------------- Kernel 2: attention (QK^T, softmax, P@V) ----------------
// Grid: (n*H)*3 row-blocks of 64 rows. Block: 128 threads (4 waves), one 16-row tile each.
__global__ void __launch_bounds__(128) attn_kernel(
    const __bf16* __restrict__ Qb, const __bf16* __restrict__ Kb,
    const __bf16* __restrict__ Vt, __bf16* __restrict__ Ob) {
  __shared__ float sc[4][16][S_DIM];   // per-wave score slab (fp32)
  __shared__ float rsum[4][16];

  const int nh   = blockIdx.x / 3;   // n*H + h
  const int rb   = blockIdx.x % 3;
  const int n    = nh >> 3;
  const int h    = nh & 7;
  const int wave = threadIdx.x >> 5;
  const int lane = threadIdx.x & 31;
  const int s0   = rb * 64 + wave * 16;
  const float scale = 0.17677669529663687f;   // 1/sqrt(32)

  const int rloc = (lane >> 4) << 3;
  const int cl   = lane & 15;

  // QK^T: D=32 is exactly one 16x16x32 WMMA K-step.
  v16bf aq = load_a_bf16(Qb + (nh * S_DIM + s0) * D_DIM, D_DIM, lane);
#pragma unroll 3
  for (int tt = 0; tt < 12; ++tt) {
    v16bf bk = load_b_bf16(Kb + (nh * S_DIM + tt * 16) * D_DIM, D_DIM, lane);
    v8f s = {};
    s = wmma_bf16(aq, bk, s);
#pragma unroll
    for (int v = 0; v < 8; ++v) sc[wave][rloc + v][tt * 16 + cl] = s[v] * scale;
  }
  __syncthreads();

  // Softmax with all 32 lanes: lane pair (row, row+16) splits the 192 columns
  // (96 each); partials combined via lane-permute shuffles.
  {
    const int row  = lane & 15;
    const int half = lane >> 4;
    float* rp = sc[wave][row] + half * 96;
    float m = -1e30f;
#pragma unroll 4
    for (int t = 0; t < 96; ++t) m = fmaxf(m, rp[t]);
    m = fmaxf(m, __shfl_xor(m, 16));
    float ssum = 0.f;
#pragma unroll 4
    for (int t = 0; t < 96; ++t) { const float e = __expf(rp[t] - m); rp[t] = e; ssum += e; }
    ssum += __shfl_xor(ssum, 16);
    if (half == 0) rsum[wave][row] = ssum;     // unnormalized; folded into P@V epilogue
  }
  __syncthreads();

  // P@V: out tile 16x32 = sum over 6 K-chunks of 32 t-values.
#pragma unroll
  for (int dt = 0; dt < 2; ++dt) {
    v8f acco = {};
#pragma unroll 2
    for (int kk = 0; kk < 6; ++kk) {
      v16bf bv = load_b_bf16(Vt + (nh * D_DIM + dt * 16) * S_DIM + kk * 32, S_DIM, lane);
      const float* pa = &sc[wave][cl][kk * 32 + rloc];
      v16bf ap;
#pragma unroll
      for (int i = 0; i < 8; ++i) ap[i]     = (__bf16)pa[i];
#pragma unroll
      for (int i = 0; i < 8; ++i) ap[8 + i] = (__bf16)pa[16 + i];
      acco = wmma_bf16(ap, bv, acco);
    }
#pragma unroll
    for (int v = 0; v < 8; ++v) {
      const int s = s0 + rloc + v;
      const int d = dt * 16 + cl;
      const float val = acco[v] / rsum[wave][rloc + v];
      Ob[(s * N_DIM + n) * C_DIM + h * D_DIM + d] = (__bf16)val;   // layout (s, n, c)
    }
  }
}

// ---------------- Kernel 3: output proj + residual + LayerNorm ----------------
// Flattened rows m = s*N + n (49152). Grid: 3072 16-row tiles. Block 256 (8 waves x 32 cols).
__global__ void __launch_bounds__(256) proj_ln_kernel(
    const __bf16* __restrict__ Ob, const __bf16* __restrict__ wo, const float* __restrict__ bo,
    const float* __restrict__ msa, const float* __restrict__ gamma, const float* __restrict__ beta,
    float* __restrict__ out) {
  __shared__ float xs[16][C_DIM];
  __shared__ float part[16][16], part2[16][16];
  __shared__ float smu[16], srs[16];

  const int r0   = blockIdx.x * 16;
  const int wave = threadIdx.x >> 5;
  const int lane = threadIdx.x & 31;

  v8f acc[2] = {};
#pragma unroll 2
  for (int k0 = 0; k0 < C_DIM; k0 += 32) {
    v16bf a = load_a_bf16(Ob + r0 * C_DIM + k0, C_DIM, lane);
    if (k0 + 32 < C_DIM)
      __builtin_prefetch(Ob + r0 * C_DIM + k0 + 32 + (lane & 15) * C_DIM, 0, 1);
    v16bf b[2];
#pragma unroll
    for (int i = 0; i < 2; ++i)
      b[i] = load_b_bf16(wo + (wave * 32 + i * 16) * C_DIM + k0, C_DIM, lane);
#pragma unroll
    for (int i = 0; i < 2; ++i)
      acc[i] = wmma_bf16(a, b[i], acc[i]);
  }

  const int rloc = (lane >> 4) << 3;
  const int cl   = lane & 15;
#pragma unroll
  for (int i = 0; i < 2; ++i) {
    const int col = wave * 32 + i * 16 + cl;
    const float bias = bo[col];
#pragma unroll
    for (int v = 0; v < 8; ++v) {
      const int row = rloc + v;
      xs[row][col] = acc[i][v] + bias + msa[(r0 + row) * C_DIM + col];   // + residual
    }
  }
  __syncthreads();

  // Parallel per-row mean/var: 256 threads -> 16 partials per row -> 16-way finish.
  {
    const int row = threadIdx.x >> 4;
    const int seg = threadIdx.x & 15;
    float s = 0.f, s2 = 0.f;
#pragma unroll
    for (int i = 0; i < 16; ++i) {
      const float x = xs[row][seg * 16 + i];
      s += x; s2 += x * x;
    }
    part[row][seg] = s;
    part2[row][seg] = s2;
  }
  __syncthreads();
  if (threadIdx.x < 16) {
    const int row = threadIdx.x;
    float s = 0.f, s2 = 0.f;
#pragma unroll
    for (int i = 0; i < 16; ++i) { s += part[row][i]; s2 += part2[row][i]; }
    const float mu  = s * (1.f / C_DIM);
    const float var = s2 * (1.f / C_DIM) - mu * mu;
    smu[row] = mu;
    srs[row] = rsqrtf(var + 1e-5f);
  }
  __syncthreads();

  {
    const int row = threadIdx.x >> 4;
    const int c0  = (threadIdx.x & 15) * 16;
    const float mu = smu[row], rs = srs[row];
#pragma unroll
    for (int i = 0; i < 16; ++i) {
      const int c = c0 + i;
      out[(r0 + row) * C_DIM + c] = (xs[row][c] - mu) * rs * gamma[c] + beta[c];
    }
  }
}

extern "C" void kernel_launch(void* const* d_in, const int* in_sizes, int n_in,
                              void* d_out, int out_size, void* d_ws, size_t ws_size,
                              hipStream_t stream) {
  const float* msa   = (const float*)d_in[0];
  const float* w_qkv = (const float*)d_in[1];
  const float* b_qkv = (const float*)d_in[2];
  const float* w_out = (const float*)d_in[3];
  const float* b_out = (const float*)d_in[4];
  const float* gamma = (const float*)d_in[5];
  const float* beta  = (const float*)d_in[6];
  float* out = (float*)d_out;

  // Workspace carving (bf16 intermediates), 256B aligned.
  auto align256 = [](size_t x) { return (x + 255) & ~size_t(255); };
  const size_t sz_wq  = align256((size_t)3 * C_DIM * C_DIM * sizeof(__bf16));
  const size_t sz_wo  = align256((size_t)C_DIM * C_DIM * sizeof(__bf16));
  const size_t sz_big = align256((size_t)N_DIM * H_DIM * S_DIM * D_DIM * sizeof(__bf16));

  uint8_t* ws = (uint8_t*)d_ws;
  __bf16* wq_bf = (__bf16*)ws;               ws += sz_wq;
  __bf16* wo_bf = (__bf16*)ws;               ws += sz_wo;
  __bf16* Qb    = (__bf16*)ws;               ws += sz_big;
  __bf16* Kb    = (__bf16*)ws;               ws += sz_big;
  __bf16* Vt    = (__bf16*)ws;               ws += sz_big;
  __bf16* Ob    = (__bf16*)ws;               ws += sz_big;
  (void)ws_size; (void)in_sizes; (void)n_in; (void)out_size;

  convert_weights_kernel<<<(3 * C_DIM * C_DIM + 255) / 256, 256, 0, stream>>>(
      w_qkv, w_out, wq_bf, wo_bf);

  qkv_kernel<<<N_DIM * (S_DIM / 16), 256, 0, stream>>>(msa, wq_bf, b_qkv, Qb, Kb, Vt);

  attn_kernel<<<N_DIM * H_DIM * 3, 128, 0, stream>>>(Qb, Kb, Vt, Ob);

  proj_ln_kernel<<<(S_DIM * N_DIM) / 16, 256, 0, stream>>>(
      Ob, wo_bf, b_out, msa, gamma, beta, out);
}
